// MultiHeadAttention_70772471103935
// MI455X (gfx1250) — matrix operand order repaired and checked by reading, not compiled, not run on hardware
//
#include <hip/hip_runtime.h>

// ---------------------------------------------------------------------------
// MI455X (gfx1250) fused MHA layer: QKV proj -> flash attention (causal +
// ALiBi) -> out proj.  Matmuls on v_wmma_f32_16x16x32_bf16 (fp32 accum).
// GEMM weight tiles staged block-wide via async global->LDS (ASYNCcnt) with
// double buffering.  B=2, S=2048, D=1024, H=16, hd=64.
// ---------------------------------------------------------------------------

typedef __bf16 v16bf  __attribute__((ext_vector_type(16)));
typedef __bf16 bf16x8 __attribute__((ext_vector_type(8)));
typedef float  v8f    __attribute__((ext_vector_type(8)));
typedef float  f32x8  __attribute__((ext_vector_type(8)));
typedef int    gv4i   __attribute__((vector_size(16)));   // matches builtin param

#define WMMA_BF16(a, b, c) \
  __builtin_amdgcn_wmma_f32_16x16x32_bf16(false, (a), false, (b), (short)0, (c), false, false)

#define BATCH 2
#define SEQ   2048
#define DIM   1024
#define NH    16
#define HD    64
#define M_TOT (BATCH * SEQ)          // 4096
#define N_QKV (3 * DIM)              // 3072

#if __has_builtin(__builtin_amdgcn_global_load_async_to_lds_b128) && \
    __has_builtin(__builtin_amdgcn_s_wait_asynccnt)
#define ASYNC_LDS 1
#else
#define ASYNC_LDS 0
#endif

// 16-byte global -> LDS copy. Async (ASYNCcnt-tracked, no VGPR round trip)
// when the toolchain exposes the gfx1250 builtin, else VGPR bounce.
__device__ __forceinline__ void copy16_to_lds(const void* g, void* l) {
#if ASYNC_LDS
  __builtin_amdgcn_global_load_async_to_lds_b128(
      (__attribute__((address_space(1))) gv4i*)(g),
      (__attribute__((address_space(3))) gv4i*)(l), 0, 0);
#else
  *(uint4*)l = *(const uint4*)g;
#endif
}

__device__ __forceinline__ void wait_async_copies() {
#if ASYNC_LDS
  __builtin_amdgcn_s_wait_asynccnt(0);
#endif
  // Fallback path: visibility comes from the __syncthreads() fence.
}

// A-fragment (16x32, bf16) per ISA 7.12.2: lane = (half, m), element e maps
// to k = 8*half + (e & 7) + 16*(e >> 3).
__device__ __forceinline__ v16bf load_a_frag_f32(const float* __restrict__ row,
                                                 int kb, int half) {
  const f32x8 r0 = *(const f32x8*)(row + kb + half * 8);
  const f32x8 r1 = *(const f32x8*)(row + kb + half * 8 + 16);
  v16bf a;
#pragma unroll
  for (int e = 0; e < 8; ++e) {
    a[e]     = (__bf16)r0[e];
    a[8 + e] = (__bf16)r1[e];
  }
  return a;
}

__device__ __forceinline__ v16bf load_a_frag_bf16(const __bf16* __restrict__ row,
                                                  int kb, int half) {
  const bf16x8 r0 = *(const bf16x8*)(row + kb + half * 8);
  const bf16x8 r1 = *(const bf16x8*)(row + kb + half * 8 + 16);
  v16bf a;
#pragma unroll
  for (int e = 0; e < 8; ++e) {
    a[e]     = r0[e];
    a[8 + e] = r1[e];
  }
  return a;
}

// ---------------------------------------------------------------------------
// Kernel 0: transpose + bf16-convert weights so WMMA B-fragments
// (lane = N column, elements = 16 k values) read from K-major [K x N] LDS tiles.
// ---------------------------------------------------------------------------
__global__ __launch_bounds__(256) void prep_kernel(const float* __restrict__ Wp,
                                                   const float* __restrict__ Wo,
                                                   __bf16* __restrict__ Wtp,
                                                   __bf16* __restrict__ Wto) {
  const size_t NP = (size_t)DIM * N_QKV;
  size_t i = (size_t)blockIdx.x * 256 + threadIdx.x;
  if (i < NP) {
    size_t d = i / N_QKV, m = i % N_QKV;
    Wtp[i] = (__bf16)Wp[m * DIM + d];
  } else {
    size_t j = i - NP;
    size_t d = j / DIM, n = j % DIM;
    Wto[j] = (__bf16)Wo[n * DIM + d];
  }
}

// ---------------------------------------------------------------------------
// Kernel 1: qkv = x @ W_packed^T + b_packed, scattered head-major:
//   Q [BH,S,hd], Kt [BH,hd,S] (transposed), V [BH,S,hd], all bf16.
// Block: 256 thr = 8 waves; block tile 128(M)x64(N); the 32x64 bf16 weight
// tile per K-step (4 KB) is async-staged to LDS ONCE per block (vs 8x from
// L2), double-buffered.  Grid: (32, 48).
// ---------------------------------------------------------------------------
__global__ __launch_bounds__(256) void qkv_kernel(const float* __restrict__ x,
                                                  const __bf16* __restrict__ Wt,
                                                  const float* __restrict__ bias,
                                                  __bf16* __restrict__ Qb,
                                                  __bf16* __restrict__ Ktb,
                                                  __bf16* __restrict__ Vb) {
  __shared__ __align__(16) __bf16 wtile[2][32 * 64];
  const int tid  = threadIdx.x;
  const int lane = tid & 31;
  const int wave = tid >> 5;
  const int half = lane >> 4;
  const int l16  = lane & 15;
  const int mb   = blockIdx.x * 128 + wave * 16;
  const int nb   = blockIdx.y * 64;

  // Each of the 256 threads stages one 16B chunk of the 32x64 tile.
  const int crow = tid >> 3;          // 0..31 (k within tile)
  const int ccol = (tid & 7) * 8;     // bf16 col offset, 16B chunks

  const float* arow = x + (size_t)(mb + l16) * DIM;
  v8f acc[4] = {};

  copy16_to_lds(Wt + (size_t)crow * N_QKV + nb + ccol, &wtile[0][crow * 64 + ccol]);

  for (int kb = 0; kb < DIM; kb += 32) {
    const int cur = (kb >> 5) & 1;
    wait_async_copies();
    __syncthreads();
    if (kb + 32 < DIM) {
      copy16_to_lds(Wt + (size_t)(kb + 32 + crow) * N_QKV + nb + ccol,
                    &wtile[cur ^ 1][crow * 64 + ccol]);
    }
    v16bf a = load_a_frag_f32(arow, kb, half);
    const __bf16* bp = &wtile[cur][l16];
#pragma unroll
    for (int t = 0; t < 4; ++t) {
      v16bf bf;
      for (int s = 0; s < 16; ++s) bf[s] = bp[(8 * half + (s & 7) + 16 * (s >> 3)) * 64 + t * 16];
      acc[t] = WMMA_BF16(a, bf, acc[t]);
    }
  }

  // Epilogue: bias + route. C layout: lane n = l16, row = r + 8*half.
#pragma unroll
  for (int t = 0; t < 4; ++t) {
    const int ncol  = nb + t * 16 + l16;
    const float bv  = bias[ncol];
    const int which = ncol >> 10;      // 0=q 1=k 2=v (uniform across lanes)
    const int d     = ncol & (DIM - 1);
    const int h     = d >> 6;
    const int dd    = d & (HD - 1);
#pragma unroll
    for (int r = 0; r < 8; ++r) {
      const int row = mb + r + 8 * half;
      const int b   = row >> 11;
      const int s   = row & (SEQ - 1);
      const int bh  = b * NH + h;
      const __bf16 val = (__bf16)(acc[t][r] + bv);
      if (which == 0)      Qb [((size_t)bh * SEQ + s) * HD + dd] = val;
      else if (which == 1) Ktb[((size_t)bh * HD + dd) * SEQ + s] = val;
      else                 Vb [((size_t)bh * SEQ + s) * HD + dd] = val;
    }
  }
}

// ---------------------------------------------------------------------------
// Kernel 2: flash attention. One wave per 16-row q tile, kv tiles of 32.
// 4 WMMA for S=Q*K^T, fp32 online softmax (causal+ALiBi) with shfl_xor row
// reductions, P transposed C-layout->A-layout through per-wave LDS
// (s_wait_dscnt fence), 4 WMMA for O += P*V.  Grid: 1024 x 128 thr.
// ---------------------------------------------------------------------------
__global__ __launch_bounds__(128) void attn_kernel(const __bf16* __restrict__ Qb,
                                                   const __bf16* __restrict__ Ktb,
                                                   const __bf16* __restrict__ Vb,
                                                   __bf16* __restrict__ Ob) {
  __shared__ bf16x8 ldsP[4][64];            // per-wave 16x32 bf16 P tile
  const int warp = threadIdx.x >> 5;
  const int lane = threadIdx.x & 31;
  const int half = lane >> 4;
  const int l16  = lane & 15;

  const int w  = blockIdx.x * 4 + warp;
  const int bh = w >> 7;
  const int qt = w & 127;
  const int qb = qt * 16;
  const int h  = bh & (NH - 1);
  const float slope = exp2f(-0.5f * (float)(h + 1));
  const float scale = 0.125f;               // 1/sqrt(64)

  const __bf16* qrow = Qb + ((size_t)bh * SEQ + qb + l16) * HD;
  v16bf qa[2];
  qa[0] = load_a_frag_bf16(qrow, 0, half);
  qa[1] = load_a_frag_bf16(qrow, 32, half);

  float run_max[8], run_sum[8];
  v8f o[4] = {};
#pragma unroll
  for (int r = 0; r < 8; ++r) { run_max[r] = -3.0e38f; run_sum[r] = 0.0f; }

  __bf16* myP = (__bf16*)&ldsP[warp][0];

  const int jmax = (qb + 15) >> 5;
  for (int j = 0; j <= jmax; ++j) {
    const int kvb = j * 32;
    v8f s0 = {}, s1 = {};
#pragma unroll
    for (int c = 0; c < 2; ++c) {
      const __bf16* kp = Ktb + ((size_t)bh * HD + c * 32) * SEQ + kvb + l16;
      v16bf b0, b1;
      for (int s = 0; s < 16; ++s) {
        const size_t ko = (size_t)(8 * half + (s & 7) + 16 * (s >> 3)) * SEQ;
        b0[s] = kp[ko];
        b1[s] = kp[ko + 16];
      }
      s0 = WMMA_BF16(qa[c], b0, s0);
      s1 = WMMA_BF16(qa[c], b1, s1);
    }
    float p0[8], p1[8];
#pragma unroll
    for (int r = 0; r < 8; ++r) {
      const int q   = qb + r + 8 * half;
      const int kv0 = kvb + l16;
      const int kv1 = kv0 + 16;
      float sv0 = (kv0 <= q) ? (s0[r] * scale + slope * (float)(kv0 - q)) : -1.0e30f;
      float sv1 = (kv1 <= q) ? (s1[r] * scale + slope * (float)(kv1 - q)) : -1.0e30f;
      float tm = fmaxf(sv0, sv1);
      tm = fmaxf(tm, __shfl_xor(tm, 1, 32));
      tm = fmaxf(tm, __shfl_xor(tm, 2, 32));
      tm = fmaxf(tm, __shfl_xor(tm, 4, 32));
      tm = fmaxf(tm, __shfl_xor(tm, 8, 32));
      const float nm   = fmaxf(run_max[r], tm);
      const float corr = __expf(run_max[r] - nm);
      const float e0 = __expf(sv0 - nm);
      const float e1 = __expf(sv1 - nm);
      float rs = e0 + e1;
      rs += __shfl_xor(rs, 1, 32);
      rs += __shfl_xor(rs, 2, 32);
      rs += __shfl_xor(rs, 4, 32);
      rs += __shfl_xor(rs, 8, 32);
      run_sum[r] = run_sum[r] * corr + rs;
      run_max[r] = nm;
      p0[r] = e0;
      p1[r] = e1;
#pragma unroll
      for (int t = 0; t < 4; ++t) o[t][r] *= corr;
    }
    // C-layout -> A-layout transpose of P through LDS.
#pragma unroll
    for (int r = 0; r < 8; ++r) {
      myP[(r + 8 * half) * 32 + l16]      = (__bf16)p0[r];
      myP[(r + 8 * half) * 32 + l16 + 16] = (__bf16)p1[r];
    }
    asm volatile("s_wait_dscnt 0" ::: "memory");
    v16bf pa;
    {
      const bf16x8 r0 = *(const bf16x8*)(myP + l16 * 32 + half * 8);
      const bf16x8 r1 = *(const bf16x8*)(myP + l16 * 32 + half * 8 + 16);
#pragma unroll
      for (int e = 0; e < 8; ++e) { pa[e] = r0[e]; pa[8 + e] = r1[e]; }
    }
    const __bf16* vp = Vb + ((size_t)bh * SEQ + kvb) * HD + l16;
#pragma unroll
    for (int t = 0; t < 4; ++t) {
      v16bf bv;
      for (int s = 0; s < 16; ++s) bv[s] = vp[(8 * half + (s & 7) + 16 * (s >> 3)) * HD + t * 16];
      o[t] = WMMA_BF16(pa, bv, o[t]);
    }
  }

  const int b = bh >> 4;
#pragma unroll
  for (int t = 0; t < 4; ++t) {
#pragma unroll
    for (int r = 0; r < 8; ++r) {
      const int q = qb + r + 8 * half;
      const float val = o[t][r] / run_sum[r];
      Ob[((size_t)(b * SEQ + q)) * DIM + h * HD + t * 16 + l16] = (__bf16)val;
    }
  }
}

// ---------------------------------------------------------------------------
// Kernel 3: out = attn @ W_out^T + b_out (fp32).  Same async-LDS staged
// tiling as kernel 1.  Grid: (32, 16), 256 thr.
// ---------------------------------------------------------------------------
__global__ __launch_bounds__(256) void outproj_kernel(const __bf16* __restrict__ A,
                                                      const __bf16* __restrict__ Wt,
                                                      const float* __restrict__ bias,
                                                      float* __restrict__ out) {
  __shared__ __align__(16) __bf16 wtile[2][32 * 64];
  const int tid  = threadIdx.x;
  const int lane = tid & 31;
  const int wave = tid >> 5;
  const int half = lane >> 4;
  const int l16  = lane & 15;
  const int mb   = blockIdx.x * 128 + wave * 16;
  const int nb   = blockIdx.y * 64;
  const int crow = tid >> 3;
  const int ccol = (tid & 7) * 8;

  const __bf16* arow = A + (size_t)(mb + l16) * DIM;
  v8f acc[4] = {};

  copy16_to_lds(Wt + (size_t)crow * DIM + nb + ccol, &wtile[0][crow * 64 + ccol]);

  for (int kb = 0; kb < DIM; kb += 32) {
    const int cur = (kb >> 5) & 1;
    wait_async_copies();
    __syncthreads();
    if (kb + 32 < DIM) {
      copy16_to_lds(Wt + (size_t)(kb + 32 + crow) * DIM + nb + ccol,
                    &wtile[cur ^ 1][crow * 64 + ccol]);
    }
    v16bf a = load_a_frag_bf16(arow, kb, half);
    const __bf16* bp = &wtile[cur][l16];
#pragma unroll
    for (int t = 0; t < 4; ++t) {
      v16bf bf;
      for (int s = 0; s < 16; ++s) bf[s] = bp[(8 * half + (s & 7) + 16 * (s >> 3)) * 64 + t * 16];
      acc[t] = WMMA_BF16(a, bf, acc[t]);
    }
  }
#pragma unroll
  for (int t = 0; t < 4; ++t) {
    const int ncol = nb + t * 16 + l16;
    const float bv = bias[ncol];
#pragma unroll
    for (int r = 0; r < 8; ++r) {
      const int row = mb + r + 8 * half;
      out[(size_t)row * DIM + ncol] = acc[t][r] + bv;
    }
  }
}

// ---------------------------------------------------------------------------
// Workspace layout (bytes):
//   0         Wt_packed  bf16 [1024 x 3072]    6291456
//   6291456   Wt_out     bf16 [1024 x 1024]    2097152
//   8388608   Q          bf16 [32 x 2048 x 64] 8388608
//   16777216  Kt         bf16 [32 x 64 x 2048] 8388608
//   25165824  V          bf16 [32 x 2048 x 64] 8388608
//   33554432  attn_out   bf16 [4096 x 1024]    8388608
//   total: 41943040
// ---------------------------------------------------------------------------
extern "C" void kernel_launch(void* const* d_in, const int* in_sizes, int n_in,
                              void* d_out, int out_size, void* d_ws, size_t ws_size,
                              hipStream_t stream) {
  (void)in_sizes; (void)n_in; (void)out_size; (void)ws_size;
  const float* x  = (const float*)d_in[0];
  const float* Wp = (const float*)d_in[1];
  const float* bp = (const float*)d_in[2];
  const float* Wo = (const float*)d_in[3];
  const float* bo = (const float*)d_in[4];
  float* out = (float*)d_out;

  char* ws = (char*)d_ws;
  __bf16* Wtp = (__bf16*)(ws + 0);
  __bf16* Wto = (__bf16*)(ws + 6291456);
  __bf16* Qb  = (__bf16*)(ws + 8388608);
  __bf16* Ktb = (__bf16*)(ws + 16777216);
  __bf16* Vb  = (__bf16*)(ws + 25165824);
  __bf16* Ob  = (__bf16*)(ws + 33554432);

  prep_kernel<<<16384, 256, 0, stream>>>(Wp, Wo, Wtp, Wto);
  qkv_kernel<<<dim3(32, 48), 256, 0, stream>>>(x, Wtp, bp, Qb, Ktb, Vb);
  attn_kernel<<<1024, 128, 0, stream>>>(Qb, Ktb, Vb, Ob);
  outproj_kernel<<<dim3(32, 16), 256, 0, stream>>>(Ob, Wto, bo, out);
}
